// MultiMamba_83897891160359
// MI455X (gfx1250) — matrix operand order repaired and checked by reading, not compiled, not run on hardware
//
#include <hip/hip_runtime.h>
#include <hip/hip_bf16.h>

// ---------------- problem constants (from reference) ----------------
#define BB   2
#define HH   28
#define LL   784          // 28*28
#define DMv  384
#define DINv 768          // DM*EXPAND
#define D2v  1536         // 2*DIN
#define DSv  16
#define DTRv 24
#define K4   4
#define EPAD 64           // DTR+2*DS = 56 padded to 64

typedef __attribute__((ext_vector_type(16))) _Float16 v16h;
typedef __attribute__((ext_vector_type(8)))  _Float16 v8h;
typedef __attribute__((ext_vector_type(8)))  float    v8f;

// scanned index j -> original L index, per direction k
__device__ __forceinline__ int orig_pos(int k, int j) {
    if (k == 0) return j;
    if (k == 1) { int h = j % 28, w = j / 28; return h * 28 + w; }
    int jj = (k == 3) ? (LL - 1 - j) : j;
    int in = jj % 7, i = (jj / 7) % 7, wg = (jj / 49) % 4, hg = jj / 196;
    return (hg * 7 + i) * 28 + (wg * 7 + in);
}

// fetch one K-step worth of fragments: A (16x32) + four B tiles (32x16 each)
__device__ __forceinline__ void load_frags(const _Float16* __restrict__ pA,
                                           const _Float16* __restrict__ pB,
                                           int Kd, int kt, int half,
                                           v16h& a, v16h b[4]) {
    // A fragment: elem i -> k = half*16+i (contiguous 32B)
    a = *(const v16h*)(pA + kt + half * 16);
    // B fragments: elem i -> k = half*16+i, col n = t*16+ln (contiguous 32B)
    #pragma unroll
    for (int t = 0; t < 4; ++t)
        b[t] = *(const v16h*)(pB + (size_t)t * 16 * Kd + kt + half * 16);
}

// ------ WMMA GEMM, 16x64 strip per wave: C[M,N] = A[M,K] * W[N,K]^T ------
// A,W row-major f16 (K-contiguous); C row-major f32, ldC = N (mult of 64).
// Software-pipelined: fragments for kt+32 load while the 4 wmmas of kt run.
__global__ void k_wmma_gemm4(const _Float16* __restrict__ A,
                             const _Float16* __restrict__ W,
                             float* __restrict__ C,
                             int M, int N, int Kd) {
    const int wavesPerBlock = blockDim.x >> 5;
    const int wid  = blockIdx.x * wavesPerBlock + (threadIdx.x >> 5);
    const int lane = threadIdx.x & 31;
    const int nq4  = N >> 6;                    // 64-wide strips
    const int mt   = wid / nq4;
    const int nq   = wid - mt * nq4;
    if (mt * 16 >= M) return;                   // whole wave exits together

    const int half = lane >> 4;                 // 0 | 1
    const int ln   = lane & 15;
    const _Float16* pA = A + (size_t)(mt * 16 + ln) * Kd;
    const _Float16* pB = W + (size_t)(nq * 64 + ln) * Kd;   // tile t: +t*16*Kd

    v8f acc[4] = {{}, {}, {}, {}};
    v16h a_cur, a_nxt;
    v16h b_cur[4], b_nxt[4];
    load_frags(pA, pB, Kd, 0, half, a_cur, b_cur);

    for (int kt = 0; kt < Kd; kt += 32) {
        // prefetch+preload next K-step (clamped on last iteration: reload,
        // branch-free so EXEC stays uniform through the WMMAs)
        const int ktn = (kt + 32 < Kd) ? (kt + 32) : kt;
        __builtin_prefetch(pA + ktn + 32, 0, 3);            // global_prefetch_b8
        __builtin_prefetch(pB + ktn + 32, 0, 3);
        load_frags(pA, pB, Kd, ktn, half, a_nxt, b_nxt);

        // 4 wmmas on the CURRENT fragments overlap the loads above
        #pragma unroll
        for (int t = 0; t < 4; ++t)
            acc[t] = __builtin_amdgcn_wmma_f32_16x16x32_f16(
                         false, a_cur, false, b_cur[t], (short)0, acc[t],
                         false, false);

        a_cur = a_nxt;
        #pragma unroll
        for (int t = 0; t < 4; ++t) b_cur[t] = b_nxt[t];
    }

    // C layout: VGPR r -> M = r + 8*half, lanes = N
    #pragma unroll
    for (int t = 0; t < 4; ++t) {
        float* pC = C + (size_t)(mt * 16 + half * 8) * N + nq * 64 + t * 16 + ln;
        #pragma unroll
        for (int r = 0; r < 8; ++r) pC[(size_t)r * N] = acc[t][r];
    }
}

// ---------------- pointwise / prep kernels --------------------------
__global__ void k_f32_to_f16(const float* __restrict__ s, _Float16* __restrict__ d, int n) {
    int i = blockIdx.x * blockDim.x + threadIdx.x;
    if (i < n) d[i] = (_Float16)s[i];
}

__global__ void k_zero(float* p, int n) {
    int i = blockIdx.x * blockDim.x + threadIdx.x;
    if (i < n) p[i] = 0.f;
}

// x_proj_w (4,56,768) -> f16 padded (4,64,768), rows 56..63 = 0
__global__ void k_pad_xproj(const float* __restrict__ s, _Float16* __restrict__ d) {
    int i = blockIdx.x * blockDim.x + threadIdx.x;
    if (i >= K4 * EPAD * DINv) return;
    int dd = i % DINv, e = (i / DINv) % EPAD, k = i / (DINv * EPAD);
    d[i] = (e < 56) ? (_Float16)s[((size_t)k * 56 + e) * DINv + dd] : (_Float16)0.f;
}

// depthwise causal conv(4) + SiLU along scanned order; out (k,b,j,d) f32+f16
__global__ void k_conv(const float* __restrict__ xz, const float* __restrict__ cw,
                       const float* __restrict__ cb,
                       float* __restrict__ c32, _Float16* __restrict__ c16) {
    int i = blockIdx.x * blockDim.x + threadIdx.x;
    if (i >= K4 * BB * LL * DINv) return;
    int d = i % DINv, j = (i / DINv) % LL, b = (i / (DINv * LL)) % BB, k = i / (DINv * LL * BB);
    float acc = cb[k * DINv + d];
    #pragma unroll
    for (int t = 0; t < 4; ++t) {
        int jj = j + t - 3;
        if (jj >= 0) {
            int l = orig_pos(k, jj);
            acc += cw[(size_t)(k * DINv + d) * 4 + t] * xz[((size_t)b * LL + l) * D2v + d];
        }
    }
    float v = acc / (1.f + __expf(-acc));       // SiLU
    c32[i] = v;
    c16[i] = (_Float16)v;
}

// delta = softplus(dt_w @ dt + dt_b), (k,b,j,d); x_dbl transposed (kb,j,64)
__global__ void k_delta(const float* __restrict__ xdblT, const float* __restrict__ dtw,
                        const float* __restrict__ dtb, float* __restrict__ delta) {
    int i = blockIdx.x * blockDim.x + threadIdx.x;
    if (i >= K4 * BB * LL * DINv) return;
    int d = i % DINv, j = (i / DINv) % LL, kb = i / (DINv * LL);
    const float* xr = xdblT + ((size_t)kb * LL + j) * EPAD;     // contiguous 24
    const float* w  = dtw + (size_t)((kb >> 1) * DINv + d) * DTRv;
    float acc = dtb[(kb >> 1) * DINv + d];
    #pragma unroll
    for (int r = 0; r < DTRv; ++r) acc += xr[r] * w[r];
    delta[i] = (acc > 20.f) ? acc : log1pf(__expf(acc));
}

// split Bm/Cm slices of transposed x_dbl into (k,b,j,16) (fully coalesced)
__global__ void k_bc(const float* __restrict__ xdblT,
                     float* __restrict__ bmt, float* __restrict__ cmt) {
    int i = blockIdx.x * blockDim.x + threadIdx.x;
    if (i >= K4 * BB * LL * DSv) return;
    int n = i % DSv, jk = i / DSv;                  // jk = kb*LL + j
    const float* base = xdblT + (size_t)jk * EPAD;
    bmt[i] = base[24 + n];
    cmt[i] = base[40 + n];
}

// sequential selective scan: one thread per (k,b,d); writes to ORIGINAL l
// (fuses rev_* + transpose + Dp skip + z-gate)
__global__ void k_scan(const float* __restrict__ delta, const float* __restrict__ c32,
                       const float* __restrict__ bmt, const float* __restrict__ cmt,
                       const float* __restrict__ Alog, const float* __restrict__ Dp,
                       const float* __restrict__ xz, float* __restrict__ outs) {
    int i = blockIdx.x * blockDim.x + threadIdx.x;
    if (i >= K4 * BB * DINv) return;
    int d = i % DINv, b = (i / DINv) % BB, k = i / (DINv * BB);
    float A[DSv], h[DSv];
    #pragma unroll
    for (int n = 0; n < DSv; ++n) {
        A[n] = -__expf(Alog[(size_t)(k * DINv + d) * DSv + n]);
        h[n] = 0.f;
    }
    const float Dv = Dp[k * DINv + d];
    const size_t kb = (size_t)(k * BB + b);
    for (int j = 0; j < LL; ++j) {
        size_t row = kb * LL + j;
        float dl  = delta[row * DINv + d];
        float u   = c32 [row * DINv + d];
        float dbu = dl * u;
        const float* Bp = bmt + row * DSv;
        const float* Cp = cmt + row * DSv;
        float y = 0.f;
        #pragma unroll
        for (int n = 0; n < DSv; ++n) {
            h[n] = __expf(dl * A[n]) * h[n] + dbu * Bp[n];
            y += h[n] * Cp[n];
        }
        int l = orig_pos(k, j);
        float z = xz[((size_t)b * LL + l) * D2v + DINv + d];
        float g = z / (1.f + __expf(-z));
        outs[(kb * LL + l) * DINv + d] = (y + Dv * u) * g;
    }
}

// per-row channel LayerNorm + mean over L accumulated into gacc (k,b,d)
__global__ void k_bi_g(const float* __restrict__ outs, const float* __restrict__ lng,
                       const float* __restrict__ lnb, float* __restrict__ gacc) {
    int wib = threadIdx.x >> 5, lane = threadIdx.x & 31;
    int kb = blockIdx.x / 98;
    int l  = (blockIdx.x % 98) * 8 + wib;       // 98*8 == 784, always valid
    const float* row = outs + ((size_t)kb * LL + l) * DINv;
    float s = 0.f, s2 = 0.f;
    for (int d = lane; d < DINv; d += 32) { float v = row[d]; s += v; s2 += v * v; }
    for (int o = 16; o > 0; o >>= 1) { s += __shfl_xor(s, o, 32); s2 += __shfl_xor(s2, o, 32); }
    float mu  = s * (1.f / DINv);
    float var = s2 * (1.f / DINv) - mu * mu;
    float rst = rsqrtf(var + 1e-5f);
    for (int d = lane; d < DINv; d += 32) {
        float xn = (row[d] - mu) * rst * lng[d] + lnb[d];
        atomicAdd(&gacc[kb * DINv + d], xn * (1.f / LL));
    }
}

// attn = sigmoid(cs(gelu(gr(g)))) per (k,b)
__global__ void k_bi_attn(const float* __restrict__ gacc,
                          const float* __restrict__ grw, const float* __restrict__ grb,
                          const float* __restrict__ csw, const float* __restrict__ csb,
                          float* __restrict__ attn) {
    __shared__ float g[DINv];
    __shared__ float t[96];
    int kb = blockIdx.x;
    for (int d = threadIdx.x; d < DINv; d += blockDim.x) g[d] = gacc[kb * DINv + d];
    __syncthreads();
    for (int e = threadIdx.x; e < 96; e += blockDim.x) {
        float a = grb[e];
        for (int d = 0; d < DINv; ++d) a += grw[(size_t)e * DINv + d] * g[d];
        t[e] = 0.5f * a * (1.f + erff(a * 0.70710678118f));   // exact gelu
    }
    __syncthreads();
    for (int d = threadIdx.x; d < DINv; d += blockDim.x) {
        float a = csb[d];
        for (int e = 0; e < 96; ++e) a += csw[(size_t)d * 96 + e] * t[e];
        attn[kb * DINv + d] = 1.f / (1.f + __expf(-a));
    }
}

// S16[b*L+l, d] = f16( sum_k outs[k,b,l,d] * attn[k,b,d] )
__global__ void k_sum_scale(const float* __restrict__ outs, const float* __restrict__ attn,
                            _Float16* __restrict__ S16) {
    int i = blockIdx.x * blockDim.x + threadIdx.x;
    if (i >= BB * LL * DINv) return;
    int d = i % DINv, bl = i / DINv;
    int b = bl / LL, l = bl % LL;
    float s = 0.f;
    #pragma unroll
    for (int k = 0; k < K4; ++k) {
        size_t kb = (size_t)(k * BB + b);
        s += outs[(kb * LL + l) * DINv + d] * attn[kb * DINv + d];
    }
    S16[i] = (_Float16)s;
}

// ---------------- host-side orchestration --------------------------
extern "C" void kernel_launch(void* const* d_in, const int* in_sizes, int n_in,
                              void* d_out, int out_size, void* d_ws, size_t ws_size,
                              hipStream_t stream) {
    const float* x    = (const float*)d_in[0];
    const float* Win  = (const float*)d_in[1];
    const float* cw   = (const float*)d_in[2];
    const float* cb   = (const float*)d_in[3];
    const float* Wxp  = (const float*)d_in[4];
    const float* dtw  = (const float*)d_in[5];
    const float* dtb  = (const float*)d_in[6];
    const float* Alog = (const float*)d_in[7];
    const float* Dp   = (const float*)d_in[8];
    const float* lng  = (const float*)d_in[9];
    const float* lnb  = (const float*)d_in[10];
    const float* grw  = (const float*)d_in[11];
    const float* grb  = (const float*)d_in[12];
    const float* csw  = (const float*)d_in[13];
    const float* csb  = (const float*)d_in[14];
    const float* Wout = (const float*)d_in[15];
    float* out = (float*)d_out;

    // deterministic workspace carve-up (256B aligned)
    char* base = (char*)d_ws;
    size_t off = 0;
    auto carve = [&](size_t bytes) -> char* {
        char* p = base + off;
        off += (bytes + 255) & ~(size_t)255;
        return p;
    };
    _Float16* x16   = (_Float16*)carve((size_t)BB * LL * DMv * 2);
    _Float16* Win16 = (_Float16*)carve((size_t)D2v * DMv * 2);
    _Float16* Wxp16 = (_Float16*)carve((size_t)K4 * EPAD * DINv * 2);
    _Float16* Wout16= (_Float16*)carve((size_t)DMv * DINv * 2);
    float*    xz    = (float*)   carve((size_t)BB * LL * D2v * 4);
    float*    c32   = (float*)   carve((size_t)K4 * BB * LL * DINv * 4);
    _Float16* c16   = (_Float16*)carve((size_t)K4 * BB * LL * DINv * 2);
    float*    xdblT = (float*)   carve((size_t)K4 * BB * LL * EPAD * 4);
    float*    delta = (float*)   carve((size_t)K4 * BB * LL * DINv * 4);
    float*    bmt   = (float*)   carve((size_t)K4 * BB * LL * DSv * 4);
    float*    cmt   = (float*)   carve((size_t)K4 * BB * LL * DSv * 4);
    float*    outsB = (float*)   carve((size_t)K4 * BB * LL * DINv * 4);
    float*    gacc  = (float*)   carve((size_t)K4 * BB * DINv * 4);
    float*    attn  = (float*)   carve((size_t)K4 * BB * DINv * 4);
    _Float16* S16   = (_Float16*)carve((size_t)BB * LL * DINv * 2);
    (void)ws_size; (void)n_in; (void)in_sizes; (void)out_size;

    const int T = 256;
    // precision conversion of GEMM operands
    k_f32_to_f16<<<(BB * LL * DMv + T - 1) / T, T, 0, stream>>>(x, x16, BB * LL * DMv);
    k_f32_to_f16<<<(D2v * DMv + T - 1) / T, T, 0, stream>>>(Win, Win16, D2v * DMv);
    k_f32_to_f16<<<(DMv * DINv + T - 1) / T, T, 0, stream>>>(Wout, Wout16, DMv * DINv);
    k_pad_xproj<<<(K4 * EPAD * DINv + T - 1) / T, T, 0, stream>>>(Wxp, Wxp16);
    k_zero<<<(K4 * BB * DINv + T - 1) / T, T, 0, stream>>>(gacc, K4 * BB * DINv);

    // xz = x @ in_proj_w^T : M=1568 N=1536 K=384 -> 98*24 strips / 4 waves
    k_wmma_gemm4<<<(98 * 24) / 4, 128, 0, stream>>>(x16, Win16, xz, BB * LL, D2v, DMv);

    // depthwise conv + SiLU in scanned order
    k_conv<<<(K4 * BB * LL * DINv + T - 1) / T, T, 0, stream>>>(xz, cw, cb, c32, c16);

    // x_dbl^T per (k,b): A=conv (M=784), W=x_proj pad (N=64), K=768
    // -> output (kb, j, 64): downstream delta/bc reads become contiguous
    for (int k = 0; k < K4; ++k)
        for (int b = 0; b < BB; ++b) {
            int kb = k * BB + b;
            k_wmma_gemm4<<<13, 128, 0, stream>>>(        // 49 strips, 13 blocks
                c16   + (size_t)kb * LL * DINv,
                Wxp16 + (size_t)k * EPAD * DINv,
                xdblT + (size_t)kb * LL * EPAD,
                LL, EPAD, DINv);
        }

    k_delta<<<(K4 * BB * LL * DINv + T - 1) / T, T, 0, stream>>>(xdblT, dtw, dtb, delta);
    k_bc<<<(K4 * BB * LL * DSv + T - 1) / T, T, 0, stream>>>(xdblT, bmt, cmt);

    // sequential selective scan + gating + un-permute
    k_scan<<<(K4 * BB * DINv + T - 1) / T, T, 0, stream>>>(
        delta, c32, bmt, cmt, Alog, Dp, xz, outsB);

    // biattn
    k_bi_g<<<K4 * BB * 98, 256, 0, stream>>>(outsB, lng, lnb, gacc);
    k_bi_attn<<<K4 * BB, 256, 0, stream>>>(gacc, grw, grb, csw, csb, attn);
    k_sum_scale<<<(BB * LL * DINv + T - 1) / T, T, 0, stream>>>(outsB, attn, S16);

    // out = S @ out_proj_w^T : M=1568 N=384 K=768 -> 98*6 strips / 4 waves
    k_wmma_gemm4<<<(98 * 6) / 4, 128, 0, stream>>>(S16, Wout16, out, BB * LL, DMv, DINv);
}
